// GPT2_32822140076254
// MI455X (gfx1250) — matrix operand order repaired and checked by reading, not compiled, not run on hardware
//
#include <hip/hip_runtime.h>
#include <hip/hip_bf16.h>

typedef __bf16 bf16;
typedef __bf16 v16bf __attribute__((ext_vector_type(16)));
typedef __bf16 v8bf  __attribute__((ext_vector_type(8)));
typedef float  v8f   __attribute__((ext_vector_type(8)));

#define Bn   8
#define Sn   512
#define Hn   8
#define DHn  128
#define Dn   1024
#define FFn  4096
#define Ln   8
#define DCn  128
#define ROWS (Bn * Sn)   // 4096

__device__ __forceinline__ float gelu_f(float x) {
    return 0.5f * x * (1.0f + erff(x * 0.70710678118654752f));
}

// CDNA5 async global->LDS (ASYNCcnt path), ISA §10 / §15.18.3.
// LDS destination = per-lane byte offset VGPR; global address = VGPR pair.
__device__ __forceinline__ void async_ld_b128(unsigned lds_off, const void* gaddr) {
    asm volatile("global_load_async_to_lds_b128 %0, %1, off"
                 :: "v"(lds_off), "v"(gaddr) : "memory");
}
__device__ __forceinline__ void wait_async0() {
    asm volatile("s_wait_asynccnt 0x0" ::: "memory");
}

// ---------------------------------------------------------------------------
// Async double-buffered bf16 WMMA GEMM (B in [N,K] "weights" layout):
//   C[z][M,N] = act( A[z][M,K] * B[z]^T + bias )
// Block tile 128x128x32, 8 waves in 4x2, wave tile 32x64 -> 8 WMMA / k-step.
// Tiles staged with global_load_async_to_lds_b128 into double-buffered LDS.
// ---------------------------------------------------------------------------
#define BM   128
#define BN2  128
#define BK   32
#define ASTR 40
#define BSTR 40

template<bool OUTBF16, bool DOGELU, bool HASBIAS>
__global__ __launch_bounds__(256)
void gemm_wmma_async(const bf16* __restrict__ A, const bf16* __restrict__ Bm,
                     void* __restrict__ Cv, const float* __restrict__ bias,
                     int M, int N, int K, int lda, int ldb, int ldc,
                     long long sAo, long long sAi, long long sBo, long long sBi,
                     long long sCo, long long sCi, int zdiv)
{
    __shared__ bf16 Alds[2][BM * ASTR];
    __shared__ bf16 Blds[2][BN2 * BSTR];

    const int z  = blockIdx.z;
    const int zo = z / zdiv, zi = z % zdiv;
    A  += (size_t)(zo * sAo + zi * sAi);
    Bm += (size_t)(zo * sBo + zi * sBi);
    const size_t coff = (size_t)(zo * sCo + zi * sCi);
    float* Cf = (float*)Cv + coff;
    bf16*  Cb = (bf16*)Cv + coff;

    const int tid   = threadIdx.x;
    const int lane  = tid & 31, wave = tid >> 5;
    const int wm    = (wave & 3) * 32;     // 4 waves along M
    const int wn    = (wave >> 2) * 64;    // 2 waves along N
    const int m0    = blockIdx.y * BM;
    const int n0    = blockIdx.x * BN2;
    const int lrow  = lane & 15, lhalf = lane >> 4;

    v8f acc[2][4] = {};

    // issue 2 A + 2 B async b128 per thread into buffer `buf` for K-slice `kk`
    auto stage = [&](int buf, int kk) {
        #pragma unroll
        for (int j = 0; j < 2; ++j) {
            int idx = tid * 2 + j;                 // 0..511
            int r = idx >> 2, c8 = (idx & 3) << 3; // row 0..127, k-chunk of 8
            async_ld_b128((unsigned)(size_t)&Alds[buf][r * ASTR + c8],
                          A + (size_t)(m0 + r) * lda + kk + c8);
            async_ld_b128((unsigned)(size_t)&Blds[buf][r * BSTR + c8],
                          Bm + (size_t)(n0 + r) * ldb + kk + c8);
        }
    };

    stage(0, 0);
    wait_async0();
    __syncthreads();

    int cur = 0;
    for (int k0 = 0; k0 < K; k0 += BK) {
        const int nxt = cur ^ 1;
        if (k0 + BK < K) stage(nxt, k0 + BK);   // prefetch next slice (async)

        // fragments per CDNA5 16-bit WMMA VGPR layout:
        // A lane (m=lrow, h=lhalf): K runs [8h..8h+7],[8h+16..8h+23] -> 2x b128
        v16bf afr[2], bfr[4];
        #pragma unroll
        for (int ti = 0; ti < 2; ++ti) {
            const bf16* ap = &Alds[cur][(wm + ti * 16 + lrow) * ASTR + lhalf * 8];
            v8bf lo = *(const v8bf*)ap;
            v8bf hi = *(const v8bf*)(ap + 16);
            afr[ti] = __builtin_shufflevector(lo, hi,
                        0,1,2,3,4,5,6,7,8,9,10,11,12,13,14,15);
        }
        // B lane (n=lrow, h=lhalf): K runs [8h..8h+7],[8h+16..8h+23] -> 2x b128
        #pragma unroll
        for (int tj = 0; tj < 4; ++tj) {
            const bf16* bp = &Blds[cur][(wn + tj * 16 + lrow) * BSTR + lhalf * 8];
            v8bf lo = *(const v8bf*)bp;
            v8bf hi = *(const v8bf*)(bp + 16);
            bfr[tj] = __builtin_shufflevector(lo, hi,
                        0,1,2,3,4,5,6,7,8,9,10,11,12,13,14,15);
        }
        #pragma unroll
        for (int ti = 0; ti < 2; ++ti)
            #pragma unroll
            for (int tj = 0; tj < 4; ++tj)
                acc[ti][tj] = __builtin_amdgcn_wmma_f32_16x16x32_bf16(
                    false, afr[ti], false, bfr[tj], (short)0, acc[ti][tj],
                    false, false);

        wait_async0();       // our prefetch landed
        __syncthreads();     // everyone done reading `cur`, prefetch visible
        cur = nxt;
    }

    // C/D layout: lane 0-15 -> N=lane, VGPR r -> M = r + 8*(lane>=16)
    #pragma unroll
    for (int ti = 0; ti < 2; ++ti) {
        #pragma unroll
        for (int tj = 0; tj < 4; ++tj) {
            const int n = n0 + wn + tj * 16 + lrow;
            float bv = HASBIAS ? bias[n] : 0.0f;
            #pragma unroll
            for (int r = 0; r < 8; ++r) {
                const int m = m0 + wm + ti * 16 + lhalf * 8 + r;
                float v = acc[ti][tj][r] + bv;
                if (DOGELU) v = gelu_f(v);
                const size_t off = (size_t)m * ldc + n;
                if (OUTBF16) Cb[off] = (bf16)v; else Cf[off] = v;
            }
        }
    }
}

// ---------------------------------------------------------------------------
// Synchronous WMMA GEMM for B in [K,N] layout (P*V): scatter-transpose in LDS.
// Block tile 128x64x32, 8 waves in 4x2, wave tile 32x32.
// ---------------------------------------------------------------------------
#define BNV 64

__global__ __launch_bounds__(256)
void gemm_wmma_nt(const bf16* __restrict__ A, const bf16* __restrict__ Bm,
                  bf16* __restrict__ Cb,
                  int M, int N, int K, int lda, int ldb, int ldc,
                  long long sAo, long long sAi, long long sBo, long long sBi,
                  long long sCo, long long sCi, int zdiv)
{
    __shared__ bf16 Alds[BM * ASTR];
    __shared__ bf16 Blds[BNV * BSTR];

    const int z  = blockIdx.z;
    const int zo = z / zdiv, zi = z % zdiv;
    A  += (size_t)(zo * sAo + zi * sAi);
    Bm += (size_t)(zo * sBo + zi * sBi);
    Cb += (size_t)(zo * sCo + zi * sCi);

    const int tid   = threadIdx.x;
    const int lane  = tid & 31, wave = tid >> 5;
    const int wm    = (wave & 3) * 32;
    const int wn    = (wave >> 2) * 32;
    const int m0    = blockIdx.y * BM;
    const int n0    = blockIdx.x * BNV;
    const int lrow  = lane & 15, lhalf = lane >> 4;

    v8f acc[2][2] = {};

    for (int k0 = 0; k0 < K; k0 += BK) {
        __syncthreads();
        #pragma unroll
        for (int j = 0; j < 2; ++j) {
            int idx = tid * 2 + j;
            int r = idx >> 2, c8 = (idx & 3) << 3;
            *(uint4*)(&Alds[r * ASTR + c8]) =
                *(const uint4*)(A + (size_t)(m0 + r) * lda + k0 + c8);
        }
        {   // B[K,N]: load row, scatter-transpose into [n][k] layout
            int kk = tid >> 3, n8 = (tid & 7) << 3;
            v8bf v = *(const v8bf*)(Bm + (size_t)(k0 + kk) * ldb + n0 + n8);
            #pragma unroll
            for (int j = 0; j < 8; ++j) Blds[(n8 + j) * BSTR + kk] = v[j];
        }
        __syncthreads();

        v16bf afr[2], bfr[2];
        #pragma unroll
        for (int ti = 0; ti < 2; ++ti) {
            const bf16* ap = &Alds[(wm + ti * 16 + lrow) * ASTR + lhalf * 8];
            v8bf lo = *(const v8bf*)ap;
            v8bf hi = *(const v8bf*)(ap + 16);
            afr[ti] = __builtin_shufflevector(lo, hi,
                        0,1,2,3,4,5,6,7,8,9,10,11,12,13,14,15);
        }
        #pragma unroll
        for (int tj = 0; tj < 2; ++tj) {
            const bf16* bp = &Blds[(wn + tj * 16 + lrow) * BSTR + lhalf * 8];
            v8bf lo = *(const v8bf*)bp;
            v8bf hi = *(const v8bf*)(bp + 16);
            bfr[tj] = __builtin_shufflevector(lo, hi,
                        0,1,2,3,4,5,6,7,8,9,10,11,12,13,14,15);
        }
        #pragma unroll
        for (int ti = 0; ti < 2; ++ti)
            #pragma unroll
            for (int tj = 0; tj < 2; ++tj)
                acc[ti][tj] = __builtin_amdgcn_wmma_f32_16x16x32_bf16(
                    false, afr[ti], false, bfr[tj], (short)0, acc[ti][tj],
                    false, false);
    }

    #pragma unroll
    for (int ti = 0; ti < 2; ++ti)
        #pragma unroll
        for (int tj = 0; tj < 2; ++tj) {
            const int n = n0 + wn + tj * 16 + lrow;
            #pragma unroll
            for (int r = 0; r < 8; ++r) {
                const int m = m0 + wm + ti * 16 + lhalf * 8 + r;
                Cb[(size_t)m * ldc + n] = (bf16)acc[ti][tj][r];
            }
        }
}

// ---------------------------------------------------------------------------
__global__ __launch_bounds__(256)
void embed_pe(const float* __restrict__ obs, const float* __restrict__ act,
              const float* __restrict__ w, const float* __restrict__ eb,
              float* __restrict__ tmp)
{
    const int row = blockIdx.x;
    const int col = blockIdx.y * 256 + threadIdx.x;
    const float* wr = w + (size_t)col * 80;
    const float* ob = obs + (size_t)row * 64;
    const float* ac = act + (size_t)row * 16;
    float s = eb[col];
    #pragma unroll 4
    for (int j = 0; j < 64; ++j) s += ob[j] * wr[j];
    #pragma unroll 4
    for (int j = 0; j < 16; ++j) s += ac[j] * wr[64 + j];
    s = gelu_f(s);
    const int pos = row & (Sn - 1);
    float div = __expf(-logf(10000.0f) * (float)(col & ~1) / (float)Dn);
    float ang = (float)pos * div;
    s += (col & 1) ? __cosf(ang) : __sinf(ang);
    tmp[(size_t)row * Dn + col] = s;
}

__global__ __launch_bounds__(256)
void add_ln(const float* __restrict__ xin, const float* __restrict__ res,
            const float* __restrict__ g, const float* __restrict__ b,
            float* __restrict__ xout, bf16* __restrict__ xbout)
{
    __shared__ float red[256];
    const int row = blockIdx.x, t = threadIdx.x;
    const float* xr = xin + (size_t)row * Dn;
    const float* rr = res ? res + (size_t)row * Dn : nullptr;
    float v[4]; float s = 0.0f;
    #pragma unroll
    for (int i = 0; i < 4; ++i) {
        int c = i * 256 + t;
        v[i] = xr[c] + (rr ? rr[c] : 0.0f);
        s += v[i];
    }
    red[t] = s; __syncthreads();
    for (int st = 128; st > 0; st >>= 1) { if (t < st) red[t] += red[t + st]; __syncthreads(); }
    const float mean = red[0] * (1.0f / Dn); __syncthreads();
    float vs = 0.0f;
    #pragma unroll
    for (int i = 0; i < 4; ++i) { float d = v[i] - mean; vs += d * d; }
    red[t] = vs; __syncthreads();
    for (int st = 128; st > 0; st >>= 1) { if (t < st) red[t] += red[t + st]; __syncthreads(); }
    const float rstd = rsqrtf(red[0] * (1.0f / Dn) + 1e-5f);
    #pragma unroll
    for (int i = 0; i < 4; ++i) {
        int c = i * 256 + t;
        float o = (v[i] - mean) * rstd * g[c] + b[c];
        xout[(size_t)row * Dn + c] = o;
        xbout[(size_t)row * Dn + c] = (bf16)o;
    }
}

__global__ __launch_bounds__(256)
void softmax_mask(const float* __restrict__ Sc, bf16* __restrict__ P,
                  const unsigned char* __restrict__ mask)
{
    const float SCALE = 0.08838834764831845f; // 1/sqrt(128)
    const int q = blockIdx.x, h = blockIdx.y, b = blockIdx.z, t = threadIdx.x;
    const size_t ro = (((size_t)b * Hn + h) * Sn + q) * Sn;
    const unsigned char* mrow = mask + ((size_t)h * Sn + q) * Sn;
    const int c0 = t, c1 = t + 256;
    float v0 = mrow[c0] ? Sc[ro + c0] * SCALE : -1e9f;
    float v1 = mrow[c1] ? Sc[ro + c1] * SCALE : -1e9f;
    __shared__ float red[256];
    red[t] = fmaxf(v0, v1); __syncthreads();
    for (int st = 128; st > 0; st >>= 1) { if (t < st) red[t] = fmaxf(red[t], red[t + st]); __syncthreads(); }
    const float mx = red[0]; __syncthreads();
    float e0 = __expf(v0 - mx), e1 = __expf(v1 - mx);
    red[t] = e0 + e1; __syncthreads();
    for (int st = 128; st > 0; st >>= 1) { if (t < st) red[t] += red[t + st]; __syncthreads(); }
    const float inv = 1.0f / red[0];
    P[ro + c0] = (bf16)(e0 * inv);
    P[ro + c1] = (bf16)(e1 * inv);
}

__global__ __launch_bounds__(256)
void final_fc(const float* __restrict__ x, const float* __restrict__ w,
              const float* __restrict__ bias, float* __restrict__ out)
{
    const int c = blockIdx.x, t = threadIdx.x;
    const int K = Sn * Dn;
    float acc[Bn] = {};
    const float* wr = w + (size_t)c * K;
    for (int k = t; k < K; k += 256) {
        float wv = wr[k];
        #pragma unroll
        for (int b = 0; b < Bn; ++b) acc[b] += x[(size_t)b * K + k] * wv;
    }
    __shared__ float red[256];
    for (int b = 0; b < Bn; ++b) {
        red[t] = acc[b]; __syncthreads();
        for (int st = 128; st > 0; st >>= 1) { if (t < st) red[t] += red[t + st]; __syncthreads(); }
        if (t == 0) out[b * DCn + c] = gelu_f(red[0] + bias[c]);
        __syncthreads();
    }
}

__global__ __launch_bounds__(256)
void f32_to_bf16(const float* __restrict__ in, bf16* __restrict__ out, int n)
{
    int i = blockIdx.x * 256 + threadIdx.x;
    if (i < n) out[i] = (bf16)in[i];
}

// ---------------------------------------------------------------------------
extern "C" void kernel_launch(void* const* d_in, const int* in_sizes, int n_in,
                              void* d_out, int out_size, void* d_ws, size_t ws_size,
                              hipStream_t stream)
{
    (void)in_sizes; (void)n_in; (void)out_size; (void)ws_size;

    const float* obs  = (const float*)d_in[0];
    const float* act  = (const float*)d_in[1];
    const unsigned char* mask = (const unsigned char*)d_in[2];
    const float* emw  = (const float*)d_in[3];
    const float* emb  = (const float*)d_in[4];
    const float* ln0g = (const float*)d_in[5];
    const float* ln0b = (const float*)d_in[6];
    const float* wq   = (const float*)d_in[7];
    const float* wk   = (const float*)d_in[8];
    const float* wv   = (const float*)d_in[9];
    const float* wo   = (const float*)d_in[10];
    const float* wo_bias = (const float*)d_in[11];
    const float* ln1g = (const float*)d_in[12];
    const float* ln1b = (const float*)d_in[13];
    const float* w1   = (const float*)d_in[14];
    const float* b1   = (const float*)d_in[15];
    const float* w2   = (const float*)d_in[16];
    const float* b2   = (const float*)d_in[17];
    const float* ln2g = (const float*)d_in[18];
    const float* ln2b = (const float*)d_in[19];
    const float* fcw  = (const float*)d_in[20];
    const float* fcb  = (const float*)d_in[21];

    char* p = (char*)d_ws;
    auto alloc = [&](size_t bytes) { char* r = p; p += (bytes + 255) & ~(size_t)255; return r; };
    float* x   = (float*)alloc((size_t)ROWS * Dn * 4);
    float* tmp = (float*)alloc((size_t)ROWS * Dn * 4);
    bf16* xb   = (bf16*)alloc((size_t)ROWS * Dn * 2);
    bf16* qb   = (bf16*)alloc((size_t)ROWS * Dn * 2);
    bf16* kb   = (bf16*)alloc((size_t)ROWS * Dn * 2);
    bf16* vb   = (bf16*)alloc((size_t)ROWS * Dn * 2);
    float* sc  = (float*)alloc((size_t)Bn * Hn * Sn * Sn * 4);
    bf16* pb   = (bf16*)alloc((size_t)Bn * Hn * Sn * Sn * 2);
    bf16* avb  = (bf16*)alloc((size_t)ROWS * Dn * 2);
    bf16* hb   = (bf16*)alloc((size_t)ROWS * FFn * 2);
    bf16* wqb  = (bf16*)alloc((size_t)Dn * Dn * 2);
    bf16* wkb  = (bf16*)alloc((size_t)Dn * Dn * 2);
    bf16* wvb  = (bf16*)alloc((size_t)Dn * Dn * 2);
    bf16* wob  = (bf16*)alloc((size_t)Dn * Dn * 2);
    bf16* w1b  = (bf16*)alloc((size_t)FFn * Dn * 2);
    bf16* w2b  = (bf16*)alloc((size_t)Dn * FFn * 2);

    const long long SD = (long long)Sn * Dn;     // 524288
    const long long SS = (long long)Sn * Sn;     // 262144

    // ---- embed + PE, then LN0 (writes x fp32 and xb bf16) ----
    embed_pe<<<dim3(ROWS, Dn / 256), 256, 0, stream>>>(obs, act, emw, emb, tmp);
    add_ln<<<ROWS, 256, 0, stream>>>(tmp, nullptr, ln0g, ln0b, x, xb);

    for (int l = 0; l < Ln; ++l) {
        const size_t dd = (size_t)Dn * Dn, fd = (size_t)FFn * Dn;
        f32_to_bf16<<<(dd + 255) / 256, 256, 0, stream>>>(wq + l * dd, wqb, (int)dd);
        f32_to_bf16<<<(dd + 255) / 256, 256, 0, stream>>>(wk + l * dd, wkb, (int)dd);
        f32_to_bf16<<<(dd + 255) / 256, 256, 0, stream>>>(wv + l * dd, wvb, (int)dd);
        f32_to_bf16<<<(dd + 255) / 256, 256, 0, stream>>>(wo + l * dd, wob, (int)dd);
        f32_to_bf16<<<(fd + 255) / 256, 256, 0, stream>>>(w1 + l * fd, w1b, (int)fd);
        f32_to_bf16<<<(fd + 255) / 256, 256, 0, stream>>>(w2 + l * fd, w2b, (int)fd);

        // Q/K/V projections: [4096,1024] x [1024,1024]^T -> bf16
        gemm_wmma_async<true, false, false><<<dim3(Dn / BN2, ROWS / BM, 1), 256, 0, stream>>>(
            xb, wqb, qb, nullptr, ROWS, Dn, Dn, Dn, Dn, Dn, 0, 0, 0, 0, 0, 0, 1);
        gemm_wmma_async<true, false, false><<<dim3(Dn / BN2, ROWS / BM, 1), 256, 0, stream>>>(
            xb, wkb, kb, nullptr, ROWS, Dn, Dn, Dn, Dn, Dn, 0, 0, 0, 0, 0, 0, 1);
        gemm_wmma_async<true, false, false><<<dim3(Dn / BN2, ROWS / BM, 1), 256, 0, stream>>>(
            xb, wvb, vb, nullptr, ROWS, Dn, Dn, Dn, Dn, Dn, 0, 0, 0, 0, 0, 0, 1);

        // scores[b,h] = Q_h K_h^T  (z = b*H + h, batched over 64)
        gemm_wmma_async<false, false, false><<<dim3(Sn / BN2, Sn / BM, Bn * Hn), 256, 0, stream>>>(
            qb, kb, sc, nullptr, Sn, Sn, DHn, Dn, Dn, Sn,
            SD, DHn, SD, DHn, (long long)Hn * SS, SS, Hn);

        softmax_mask<<<dim3(Sn, Hn, Bn), 256, 0, stream>>>(sc, pb, mask);

        // AV[b,h] = P V_h  (B operand is [K,N])
        gemm_wmma_nt<<<dim3(DHn / BNV, Sn / BM, Bn * Hn), 256, 0, stream>>>(
            pb, vb, avb, Sn, DHn, Sn, Sn, Dn, Dn,
            (long long)Hn * SS, SS, SD, DHn, SD, DHn, Hn);

        // output projection + residual + LN1
        gemm_wmma_async<false, false, true><<<dim3(Dn / BN2, ROWS / BM, 1), 256, 0, stream>>>(
            avb, wob, tmp, wo_bias + (size_t)l * Dn, ROWS, Dn, Dn, Dn, Dn, Dn,
            0, 0, 0, 0, 0, 0, 1);
        add_ln<<<ROWS, 256, 0, stream>>>(tmp, x, ln1g + (size_t)l * Dn, ln1b + (size_t)l * Dn, x, xb);

        // FFN
        gemm_wmma_async<true, true, true><<<dim3(FFn / BN2, ROWS / BM, 1), 256, 0, stream>>>(
            xb, w1b, hb, b1 + (size_t)l * FFn, ROWS, FFn, Dn, Dn, Dn, FFn,
            0, 0, 0, 0, 0, 0, 1);
        gemm_wmma_async<false, false, true><<<dim3(Dn / BN2, ROWS / BM, 1), 256, 0, stream>>>(
            hb, w2b, tmp, b2 + (size_t)l * Dn, ROWS, Dn, FFn, FFn, FFn, Dn,
            0, 0, 0, 0, 0, 0, 1);
        add_ln<<<ROWS, 256, 0, stream>>>(tmp, x, ln2g + (size_t)l * Dn, ln2b + (size_t)l * Dn, x, xb);
    }

    final_fc<<<DCn, 256, 0, stream>>>(x, fcw, fcb, (float*)d_out);
}
